// SelfAttention_69346541961289
// MI455X (gfx1250) — matrix operand order repaired and checked by reading, not compiled, not run on hardware
//
#include <hip/hip_runtime.h>
#include <math.h>

// ---------------------------------------------------------------------------
// Self-attention (softmax over QUERY axis), B=8, N=2048, D=768, fp32 I/O.
// bf16 WMMA (v_wmma_f32_16x16x32_bf16) for all three GEMMs, fp32 accumulate,
// materialized fp32 score matrix (134MB, fits MI455X 192MB L2).
// 2-deep software pipeline, unrolled x2 with ping-pong stage registers.
// Named accumulators (not arrays) so C=D register coalescing removes the
// per-iteration accumulator copies + WMMA->VALU hazard nops.
// ---------------------------------------------------------------------------

#define B_ 8
#define N_ 2048
#define D_ 768
#define INV_SCALE 0.10825317547305482f   // 1/sqrt(768/9)

typedef __attribute__((ext_vector_type(16))) __bf16          v16bf;
typedef __attribute__((ext_vector_type(16))) unsigned short  v16u;
typedef __attribute__((ext_vector_type(8)))  unsigned short  v8u;
typedef __attribute__((ext_vector_type(8)))  float           v8f;

// fp32 -> bf16 raw bits, round-to-nearest-even
static __device__ __forceinline__ unsigned short bf(float f) {
  unsigned u = __builtin_bit_cast(unsigned, f);
  u += 0x7FFFu + ((u >> 16) & 1u);
  return (unsigned short)(u >> 16);
}

static __device__ __forceinline__ v16u packA(v8u lo, v8u hi) {
  v16u a;
#pragma unroll
  for (int i = 0; i < 8; ++i) { a[i] = lo[i]; a[i + 8] = hi[i]; }
  return a;
}

static __device__ __forceinline__ v8f wmma_bf16(v16u a, v16u b, v8f c) {
  return __builtin_amdgcn_wmma_f32_16x16x32_bf16(
      false, __builtin_bit_cast(v16bf, a),
      false, __builtin_bit_cast(v16bf, b),
      (short)0, c, false, false);
}

// ---- shared GEMM pipeline stage (A tile + 4 B tiles) ----------------------
struct GemmStage { v8u alo, ahi; v16u b0, b1, b2, b3; };

static __device__ __forceinline__ void gemm_load(GemmStage& s,
                                                 const unsigned short* __restrict__ pa,
                                                 const unsigned short* __restrict__ pb,
                                                 int d) {
  s.alo = *(const v8u*)(pa + d);
  s.ahi = *(const v8u*)(pa + d + 8);
  s.b0  = *(const v16u*)(pb + (size_t)( 0) * D_ + d);
  s.b1  = *(const v16u*)(pb + (size_t)(16) * D_ + d);
  s.b2  = *(const v16u*)(pb + (size_t)(32) * D_ + d);
  s.b3  = *(const v16u*)(pb + (size_t)(48) * D_ + d);
}

static __device__ __forceinline__ void gemm_mma(v8f& a0, v8f& a1, v8f& a2, v8f& a3,
                                                const GemmStage& s) {
  v16u a = packA(s.alo, s.ahi);
  a0 = wmma_bf16(a, s.b0, a0);
  a1 = wmma_bf16(a, s.b1, a1);
  a2 = wmma_bf16(a, s.b2, a2);
  a3 = wmma_bf16(a, s.b3, a3);
}

// ---------------------------------------------------------------------------
// Kernel 1a: convert Wq/Wk/Wv fp32 -> bf16, packed [3][768][768]
// ---------------------------------------------------------------------------
__global__ __launch_bounds__(256) void k_cvt_w(const float* __restrict__ Wq,
                                               const float* __restrict__ Wk,
                                               const float* __restrict__ Wv,
                                               unsigned short* __restrict__ Wh) {
  int i = blockIdx.x * 256 + threadIdx.x;
  const int per = D_ * D_;
  if (i >= 3 * per) return;
  const float* src = (i < per) ? Wq : (i < 2 * per ? Wk : Wv);
  Wh[i] = bf(src[i % per]);
}

// ---------------------------------------------------------------------------
// Kernel 1b: convert x fp32 -> bf16 once (row tiles re-read 36x by k_qkv).
// ---------------------------------------------------------------------------
__global__ __launch_bounds__(256) void k_cvt_x(const float* __restrict__ x,
                                               unsigned short* __restrict__ Xh) {
  size_t i = ((size_t)blockIdx.x * 256 + threadIdx.x) * 8;
  v8f v = *(const v8f*)(x + i);
  v8u h;
#pragma unroll
  for (int j = 0; j < 8; ++j) h[j] = bf(v[j]);
  *(v8u*)(Xh + i) = h;
}

// ---------------------------------------------------------------------------
// Kernel 2: QKV projection.  y = x @ W^T + b.
// One wave = 16-row tile x 64 output cols for one of {Q,K,V}.
// Q,K row-major bf16; V stored TRANSPOSED Vt[d][key] (contiguous D-store).
// ---------------------------------------------------------------------------
__global__ __launch_bounds__(128) void k_qkv(const unsigned short* __restrict__ Xh,
                                             const unsigned short* __restrict__ Wh,
                                             const float* __restrict__ bq,
                                             const float* __restrict__ bk,
                                             const float* __restrict__ bv,
                                             unsigned short* __restrict__ Qh,
                                             unsigned short* __restrict__ Kh,
                                             unsigned short* __restrict__ Vt) {
  int wid  = blockIdx.x * 4 + (threadIdx.x >> 5);
  int lane = threadIdx.x & 31;
  int cg = wid % 12; int tmp = wid / 12;
  int m  = tmp % 3;  int r   = tmp / 3;            // r: 0..1023 (row tile over B*N)
  int row0 = r * 16, e0 = cg * 64;
  int arow = lane & 15, kh = lane >> 4;
  int koff = kh * 16, mbase = kh * 8;

  const unsigned short* pa = Xh + (size_t)(row0 + arow) * D_ + koff;
  const unsigned short* pb = Wh + ((size_t)m * D_ + e0 + arow) * D_ + kh * 16;
  v8f acc0 = {}, acc1 = {}, acc2 = {}, acc3 = {};

  GemmStage s0, s1;
  gemm_load(s0, pa, pb, 0);
  for (int d0 = 0; d0 < D_ - 64; d0 += 64) {
    gemm_load(s1, pa, pb, d0 + 32);
    gemm_mma(acc0, acc1, acc2, acc3, s0);
    gemm_load(s0, pa, pb, d0 + 64);
    gemm_mma(acc0, acc1, acc2, acc3, s1);
  }
  gemm_load(s1, pa, pb, D_ - 32);
  gemm_mma(acc0, acc1, acc2, acc3, s0);
  gemm_mma(acc0, acc1, acc2, acc3, s1);

  v8f accs[4] = {acc0, acc1, acc2, acc3};
  const float* bias = (m == 0) ? bq : (m == 1 ? bk : bv);
  if (m < 2) {
    unsigned short* dst = (m == 0) ? Qh : Kh;
#pragma unroll
    for (int t = 0; t < 4; ++t) {
      float bb = bias[e0 + t * 16 + arow];
#pragma unroll
      for (int i = 0; i < 8; ++i)
        dst[(size_t)(row0 + mbase + i) * D_ + e0 + t * 16 + arow] = bf(accs[t][i] + bb);
    }
  } else {
    int b = row0 >> 11, n0 = row0 & (N_ - 1);
#pragma unroll
    for (int t = 0; t < 4; ++t) {
      float bb = bias[e0 + t * 16 + arow];
      v8u v;
#pragma unroll
      for (int i = 0; i < 8; ++i) v[i] = bf(accs[t][i] + bb);
      *(v8u*)(Vt + (size_t)b * D_ * N_ + (size_t)(e0 + t * 16 + arow) * N_ + n0 + mbase) = v;
    }
  }
}

// ---------------------------------------------------------------------------
// Kernel 3: scores S[b][q][k] = (Q . K)/SCALE.
// Wave computes TRANSPOSED tile D[m=key][n=query] = K_tile @ Q^T so the
// row-major S store is per-lane contiguous (32B).  16 keys x 64 queries.
// ---------------------------------------------------------------------------
__global__ __launch_bounds__(128) void k_scores(const unsigned short* __restrict__ Qh,
                                                const unsigned short* __restrict__ Kh,
                                                float* __restrict__ S) {
  int wid  = blockIdx.x * 4 + (threadIdx.x >> 5);
  int lane = threadIdx.x & 31;
  int qg = wid % 32; int tmp = wid / 32;
  int kt = tmp % 128; int b = tmp / 128;
  int k0 = kt * 16, q0 = qg * 64;
  int rrow = lane & 15, kh = lane >> 4;
  int koff = kh * 16, mbase = kh * 8;

  const unsigned short* pa = Kh + ((size_t)b * N_ + k0 + rrow) * D_ + koff;
  const unsigned short* pb = Qh + ((size_t)b * N_ + q0 + rrow) * D_ + kh * 16;
  v8f acc0 = {}, acc1 = {}, acc2 = {}, acc3 = {};

  GemmStage s0, s1;
  gemm_load(s0, pa, pb, 0);
  for (int d0 = 0; d0 < D_ - 64; d0 += 64) {
    gemm_load(s1, pa, pb, d0 + 32);
    gemm_mma(acc0, acc1, acc2, acc3, s0);
    gemm_load(s0, pa, pb, d0 + 64);
    gemm_mma(acc0, acc1, acc2, acc3, s1);
  }
  gemm_load(s1, pa, pb, D_ - 32);
  gemm_mma(acc0, acc1, acc2, acc3, s0);
  gemm_mma(acc0, acc1, acc2, acc3, s1);

  v8f accs[4] = {acc0, acc1, acc2, acc3};
#pragma unroll
  for (int t = 0; t < 4; ++t) {
    float* ps = S + ((size_t)b * N_ + q0 + t * 16 + rrow) * N_ + k0 + mbase;
    v8f v;
#pragma unroll
    for (int i = 0; i < 8; ++i) v[i] = accs[t][i] * INV_SCALE;
    *(v8f*)ps = v;
  }
}

// ---------------------------------------------------------------------------
// Kernel 4: per-KEY-column softmax stats over the query axis (coalesced:
// consecutive threads -> consecutive k).  Online max/sum.
// ---------------------------------------------------------------------------
__global__ __launch_bounds__(256) void k_stats(const float* __restrict__ S,
                                               float* __restrict__ Mx,
                                               float* __restrict__ Rr) {
  int b = blockIdx.x >> 3;
  int k = ((blockIdx.x & 7) << 8) + threadIdx.x;
  const float* col = S + (size_t)b * N_ * N_ + k;
  float m = -INFINITY, s = 0.f;
#pragma unroll 4
  for (int q = 0; q < N_; ++q) {
    float v  = col[(size_t)q * N_];
    float nm = fmaxf(m, v);
    s = s * __expf(m - nm) + __expf(v - nm);
    m = nm;
  }
  Mx[b * N_ + k] = m;
  Rr[b * N_ + k] = 1.0f / s;
}

// ---- attention pipeline stage ---------------------------------------------
struct AttnStage { v8f sv0, sv1, mv0, mv1, rv0, rv1; v8u vl, vh; };

static __device__ __forceinline__ void attn_load(AttnStage& st,
                                                 const float* __restrict__ Sb,
                                                 const float* __restrict__ mb,
                                                 const float* __restrict__ rb,
                                                 const unsigned short* __restrict__ pvb,
                                                 int k) {
  st.sv0 = *(const v8f*)(Sb + k);  st.sv1 = *(const v8f*)(Sb + k + 8);
  st.mv0 = *(const v8f*)(mb + k);  st.mv1 = *(const v8f*)(mb + k + 8);
  st.rv0 = *(const v8f*)(rb + k);  st.rv1 = *(const v8f*)(rb + k + 8);
  st.vl  = *(const v8u*)(pvb + k); st.vh  = *(const v8u*)(pvb + k + 8);
}

static __device__ __forceinline__ void attn_store_w(const AttnStage& st,
                                                    unsigned short* dsts) {
  v8u w0, w1;
#pragma unroll
  for (int i = 0; i < 8; ++i) {
    w0[i] = bf(__expf(st.sv0[i] - st.mv0[i]) * st.rv0[i]);
    w1[i] = bf(__expf(st.sv1[i] - st.mv1[i]) * st.rv1[i]);
  }
  *(v8u*)dsts       = w0;
  *(v8u*)(dsts + 8) = w1;
}

static __device__ __forceinline__ void attn_mma(v8f& a0, v8f& a1, v8f& a2, v8f& a3,
                                                const AttnStage& st,
                                                const unsigned short* ldsbuf,
                                                int rrow, int kh) {
  v16u a = packA(st.vl, st.vh);
  v16u b0 = *(const v16u*)&ldsbuf[( 0 + rrow) * 32 + kh * 16];
  v16u b1 = *(const v16u*)&ldsbuf[(16 + rrow) * 32 + kh * 16];
  v16u b2 = *(const v16u*)&ldsbuf[(32 + rrow) * 32 + kh * 16];
  v16u b3 = *(const v16u*)&ldsbuf[(48 + rrow) * 32 + kh * 16];
  a0 = wmma_bf16(a, b0, a0);
  a1 = wmma_bf16(a, b1, a1);
  a2 = wmma_bf16(a, b2, a2);
  a3 = wmma_bf16(a, b3, a3);
}

// ---------------------------------------------------------------------------
// Kernel 5: out = P @ V + x,  P[q,k] = exp(S-m[k]) * r[k].
// Block = 4 waves.  Per k-step the block cooperatively builds the 64q x 32k
// bf16 weight tile in double-buffered LDS; each wave owns one 16-wide d-tile
// of Vt (A operand) and runs 4 WMMAs (B tiles via ds_load).  Ping-pong
// stages + alternating LDS buffers, one barrier per k-step.
// ---------------------------------------------------------------------------
__global__ __launch_bounds__(128) void k_attn(const float* __restrict__ S,
                                              const float* __restrict__ Mx,
                                              const float* __restrict__ Rr,
                                              const unsigned short* __restrict__ Vt,
                                              const float* __restrict__ x,
                                              float* __restrict__ out) {
  __shared__ __align__(32) unsigned short ldsw[2][64 * 32];

  int dg = blockIdx.x % 12; int tmp = blockIdx.x / 12;
  int qg = tmp % 32; int b = tmp / 32;
  int q0 = qg * 64;
  int wvid = threadIdx.x >> 5, lane = threadIdx.x & 31;
  int d0 = dg * 64 + wvid * 16;
  int rrow = lane & 15, kh = lane >> 4;
  int koff = kh * 16, mbase = kh * 8;

  // cooperative weight-tile indices: thread -> (query row, 16-wide k half)
  int cq  = threadIdx.x >> 1;
  int ckh = threadIdx.x & 1;
  const float* Sb = S  + ((size_t)b * N_ + q0 + cq) * N_ + ckh * 16;
  const float* mb = Mx + b * N_ + ckh * 16;
  const float* rb = Rr + b * N_ + ckh * 16;
  const unsigned short* pvb = Vt + ((size_t)b * D_ + d0 + rrow) * N_ + koff;
  unsigned short* dst0 = &ldsw[0][cq * 32 + ckh * 16];
  unsigned short* dst1 = &ldsw[1][cq * 32 + ckh * 16];

  v8f acc0 = {}, acc1 = {}, acc2 = {}, acc3 = {};

  AttnStage t0, t1;
  attn_load(t0, Sb, mb, rb, pvb, 0);
  for (int k0 = 0; k0 < N_ - 64; k0 += 64) {
    // stage A (buf 0): last buf0 reads were before the previous barrier
    attn_store_w(t0, dst0);
    __syncthreads();
    attn_load(t1, Sb, mb, rb, pvb, k0 + 32);
    attn_mma(acc0, acc1, acc2, acc3, t0, ldsw[0], rrow, kh);
    // stage B (buf 1)
    attn_store_w(t1, dst1);
    __syncthreads();
    attn_load(t0, Sb, mb, rb, pvb, k0 + 64);
    attn_mma(acc0, acc1, acc2, acc3, t1, ldsw[1], rrow, kh);
  }
  // tail: k-steps N_-64 and N_-32
  attn_store_w(t0, dst0);
  __syncthreads();
  attn_load(t1, Sb, mb, rb, pvb, N_ - 32);
  attn_mma(acc0, acc1, acc2, acc3, t0, ldsw[0], rrow, kh);
  attn_store_w(t1, dst1);
  __syncthreads();
  attn_mma(acc0, acc1, acc2, acc3, t1, ldsw[1], rrow, kh);

  // ---- epilogue: out^T tile -> row-major [q][d] contiguous stores, + x ----
  v8f accs[4] = {acc0, acc1, acc2, acc3};
#pragma unroll
  for (int t = 0; t < 4; ++t) {
    size_t off = ((size_t)b * N_ + q0 + t * 16 + rrow) * D_ + d0 + mbase;
    v8f xv = *(const v8f*)(x + off);
    v8f rr;
#pragma unroll
    for (int i = 0; i < 8; ++i) rr[i] = accs[t][i] + xv[i];
    *(v8f*)(out + off) = rr;
  }
}

// ---------------------------------------------------------------------------
// Host launcher
// ---------------------------------------------------------------------------
extern "C" void kernel_launch(void* const* d_in, const int* in_sizes, int n_in,
                              void* d_out, int out_size, void* d_ws, size_t ws_size,
                              hipStream_t stream) {
  const float* x  = (const float*)d_in[0];
  const float* Wq = (const float*)d_in[1];
  const float* bq = (const float*)d_in[2];
  const float* Wk = (const float*)d_in[3];
  const float* bk = (const float*)d_in[4];
  const float* Wv = (const float*)d_in[5];
  const float* bv = (const float*)d_in[6];
  float* out = (float*)d_out;

  // workspace layout (bytes)
  constexpr size_t WH_B  = (size_t)3 * D_ * D_ * 2;          //   3.5 MB
  constexpr size_t XB_B  = (size_t)B_ * N_ * D_ * 2;         //  25.2 MB
  constexpr size_t S_B   = (size_t)B_ * N_ * N_ * 4;         // 134.2 MB
  char* w = (char*)d_ws;
  unsigned short* Wh = (unsigned short*)(w);
  unsigned short* Xh = (unsigned short*)(w + WH_B);
  unsigned short* Qh = (unsigned short*)(w + WH_B + XB_B);
  unsigned short* Kh = (unsigned short*)(w + WH_B + 2 * XB_B);
  unsigned short* Vt = (unsigned short*)(w + WH_B + 3 * XB_B);
  float*          S  = (float*)(w + WH_B + 4 * XB_B);
  float*          Mx = (float*)(w + WH_B + 4 * XB_B + S_B);
  float*          Rr = Mx + (size_t)B_ * N_;
  (void)in_sizes; (void)n_in; (void)out_size; (void)ws_size;

  // 1) fp32 -> bf16 pre-passes
  k_cvt_w<<<(3 * D_ * D_) / 256, 256, 0, stream>>>(Wq, Wk, Wv, Wh);
  k_cvt_x<<<(int)((size_t)B_ * N_ * D_ / 8 / 256), 256, 0, stream>>>(x, Xh);
  // 2) QKV projection: 1024 row-tiles * 3 matrices * 12 col-groups = 36864 waves
  k_qkv<<<36864 / 4, 128, 0, stream>>>(Xh, Wh, bq, bk, bv, Qh, Kh, Vt);
  // 3) scores: 8 * 128 key-tiles * 32 query-groups = 32768 waves
  k_scores<<<32768 / 4, 128, 0, stream>>>(Qh, Kh, S);
  // 4) per-key-column softmax stats
  k_stats<<<B_ * N_ / 256, 256, 0, stream>>>(S, Mx, Rr);
  // 5) out = P @ V + x : 8 * 32 q-groups * 12 d-groups blocks of 4 waves
  k_attn<<<B_ * 32 * 12, 128, 0, stream>>>(S, Mx, Rr, Vt, x, out);
}
